// MultiheadAttention_27865747816537
// MI455X (gfx1250) — matrix operand order repaired and checked by reading, not compiled, not run on hardware
//
#include <hip/hip_runtime.h>

// ---------------------------------------------------------------------------
// Multi-head attention forward for MI455X (gfx1250), all matmuls on WMMA,
// GEMM B-tiles staged in LDS by the Tensor Data Mover (double-buffered).
// B=2, S=2048, D=1024, H=16, HD=64.  f16 data path, f32 accumulation.
// ---------------------------------------------------------------------------

typedef _Float16 half_t;
typedef __attribute__((ext_vector_type(16))) _Float16     v16h;
typedef __attribute__((ext_vector_type(8)))  _Float16     v8h;
typedef __attribute__((ext_vector_type(8)))  float        v8f;
typedef __attribute__((ext_vector_type(4)))  unsigned int v4u;
typedef __attribute__((ext_vector_type(4)))  int          v4i;
typedef __attribute__((ext_vector_type(8)))  int          v8i;

#define WMMA_F16(a, b, c) \
  __builtin_amdgcn_wmma_f32_16x16x32_f16(false, (a), false, (b), (short)0, (c), false, false)

#if defined(__gfx1250__) && __has_builtin(__builtin_amdgcn_tensor_load_to_lds) && \
    __has_builtin(__builtin_amdgcn_s_wait_tensorcnt)
#define USE_TDM 1
#else
#define USE_TDM 0
#endif

static __device__ inline v16h concat8(v8h lo, v8h hi) {
  return __builtin_shufflevector(lo, hi, 0,1,2,3,4,5,6,7,8,9,10,11,12,13,14,15);
}

// A fragment: 16x32 (MxK) tile starting at `a` (row-major, leading dim ld).
// Contiguous k order used here (matched by B): lanes 0-15 -> rows 0-15 holding K 0..15;
//             lanes 16-31 -> rows 0-15 holding K 16..31.
static __device__ inline v16h load_a_frag(const half_t* a, int ld, int lane) {
  const int r  = lane & 15;
  const int kh = (lane >> 4) << 4;                 // 0 or 16
  const half_t* p = a + (size_t)r * ld + kh;
  v8h c0 = *(const v8h*)(p);                       // K kh..kh+7
  v8h c1 = *(const v8h*)(p + 8);                   // K kh+8..kh+15
  return concat8(c0, c1);
}

// B fragment: 32x16 (KxN) tile; source is BT = Bᵀ rows (N x K row-major).
// Contiguous k order used here: lane n (0-15) -> column n, K 0..15 ; lane n+16 -> column n, K 16..31.
static __device__ inline v16h load_b_frag(const half_t* bt, int ld, int lane) {
  const int n  = lane & 15;
  const int kh = (lane >> 4) << 4;                 // 0 or 16
  const half_t* p = bt + (size_t)n * ld + kh;
  v8h c0 = *(const v8h*)(p);
  v8h c1 = *(const v8h*)(p + 8);
  return concat8(c0, c1);
}

#if USE_TDM
// Issue a TDM DMA: 64 rows x 32 f16 tile from row-major (ldK) global memory
// into LDS at byte offset lds_off, with LDS padding 32+8 f16 per row
// (pad_interval code 3 = 16 dwords, pad_amount code 3 = 4 dwords -> stride 40).
static __device__ inline void tdm_stage_bt_tile(unsigned lds_off,
                                                const half_t* gsrc, int ldK) {
  const unsigned long long ga = (unsigned long long)(uintptr_t)gsrc;
  v4u g0;
  g0[0] = 1u;                                         // count=1, user mode
  g0[1] = lds_off;                                    // lds_addr (bytes)
  g0[2] = (unsigned)(ga & 0xFFFFFFFFu);               // global_addr[31:0]
  g0[3] = (unsigned)((ga >> 32) & 0x01FFFFFFu)        // global_addr[56:32]
        | (2u << 30);                                 // type = 2 ("image")
  v8i g1;
  g1[0] = (int)((1u << 16)                            // data_size = 2 bytes
              | (1u << 20)                            // pad_enable
              | (3u << 22)                            // pad_interval: 16 dwords
              | (3u << 25));                          // pad_amount: 4 dwords
  g1[1] = (int)(((unsigned)ldK & 0xFFFFu) << 16);     // tensor_dim0 lo16
  g1[2] = (int)((((unsigned)ldK >> 16) & 0xFFFFu)     // tensor_dim0 hi16
              | (4096u << 16));                       // tensor_dim1 lo16 (rows)
  g1[3] = (int)(32u << 16);                           // dim1 hi16=0, tile_dim0=32
  g1[4] = (int)(64u);                                 // tile_dim1=64, tile_dim2=0
  g1[5] = (int)((unsigned)ldK);                       // tensor_dim0_stride lo32
  g1[6] = 0;                                          // stride hi16, dim1_stride lo
  g1[7] = 0;
  v4i g2 = {0, 0, 0, 0};
  v4i g3 = {0, 0, 0, 0};
#if __clang_major__ >= 23
  v8i g4 = {0, 0, 0, 0, 0, 0, 0, 0};
  __builtin_amdgcn_tensor_load_to_lds(g0, g1, g2, g3, g4, 0);
#else
  __builtin_amdgcn_tensor_load_to_lds(g0, g1, g2, g3, 0);
#endif
}
#endif

// ---------------------------------------------------------------------------
__global__ void mha_cvt_f32_f16(const float* __restrict__ in,
                                half_t* __restrict__ out, int n) {
  int i = blockIdx.x * blockDim.x + threadIdx.x;
  const int stride = gridDim.x * blockDim.x;
  for (; i < n; i += stride) out[i] = (half_t)in[i];
}

// ---------------------------------------------------------------------------
// C[M,N] = A[M,K] · BT[N,K]ᵀ   (torch Linear: y = x @ W.T, BT = W rows)
// Block: 256 threads = 8 waves, tile 128(M) x 64(N).
// BT tile (64x32 f16) staged into double-buffered LDS by the TDM (wave 0
// issues tensor_load_to_lds for step i+1 while all waves run WMMAs on step i),
// falling back to cooperative vector loads when the TDM builtin is absent.
// mode 0: write f16 -> [B,H,S,HD]   (Q, K)
// mode 1: write f16 -> [B,H,HD,S]   (V transposed, for contiguous P·V loads)
// mode 2: write f32 + bias -> flat [M,N]  (output projection)
__global__ __launch_bounds__(256)
void mha_gemm_nt(const half_t* __restrict__ A, const half_t* __restrict__ BT,
                 int M, int N, int K, int S, int mode,
                 half_t* __restrict__ outH, float* __restrict__ outF,
                 const float* __restrict__ bias) {
  __shared__ half_t lb[2][64 * 40];                // 2 x (64 rows x 32 K, stride 40)
  const int lane = threadIdx.x & 31;
  const int wave = threadIdx.x >> 5;               // 0..7
  const int m0 = blockIdx.x * 128 + wave * 16;
  const int n0 = blockIdx.y * 64;
  const half_t* btBase = BT + (size_t)n0 * K;

  v8f acc[4];
#pragma unroll
  for (int j = 0; j < 4; ++j) acc[j] = (v8f){0.f,0.f,0.f,0.f,0.f,0.f,0.f,0.f};

#if USE_TDM
  if (threadIdx.x == 0)
    tdm_stage_bt_tile((unsigned)(uintptr_t)&lb[0][0], btBase, K);
  if (threadIdx.x < 32) __builtin_amdgcn_s_wait_tensorcnt(0);
#else
  {
    const int t  = threadIdx.x;
    const int nn = t >> 2;
    const int kc = (t & 3) << 3;
    *(v8h*)(&lb[0][nn * 40 + kc]) =
        *(const v8h*)(&btBase[(size_t)nn * K + kc]);
  }
#endif
  __syncthreads();

  int cur = 0;
  for (int k0 = 0; k0 < K; k0 += 32) {
    const bool more = (k0 + 32) < K;
    // stage next BT tile into the other buffer while computing this one
    if (more) {
#if USE_TDM
      if (threadIdx.x == 0)
        tdm_stage_bt_tile((unsigned)(uintptr_t)&lb[cur ^ 1][0],
                          btBase + k0 + 32, K);
#else
      const int t  = threadIdx.x;
      const int nn = t >> 2;
      const int kc = (t & 3) << 3;
      *(v8h*)(&lb[cur ^ 1][nn * 40 + kc]) =
          *(const v8h*)(&btBase[(size_t)nn * K + k0 + 32 + kc]);
#endif
      // warm next A fragment while WMMAs run
      __builtin_prefetch(A + (size_t)(m0 + (lane & 15)) * K + k0 + 32, 0, 0);
    }

    v16h af = load_a_frag(A + (size_t)m0 * K + k0, K, lane);
#pragma unroll
    for (int j = 0; j < 4; ++j) {
      const int n  = lane & 15;
      const int kh = (lane >> 4) << 4;
      const half_t* p = &lb[cur][(j * 16 + n) * 40 + kh];
      v16h bf = concat8(*(const v8h*)p, *(const v8h*)(p + 8));
      acc[j] = WMMA_F16(af, bf, acc[j]);
    }

#if USE_TDM
    if (threadIdx.x < 32) __builtin_amdgcn_s_wait_tensorcnt(0);
#endif
    __syncthreads();
    cur ^= 1;
  }

  // epilogue (C layout: reg r, lane l -> row r + 8*(l>=16), col l&15)
  const int col0   = lane & 15;
  const int rowoff = (lane >> 4) << 3;
#pragma unroll
  for (int j = 0; j < 4; ++j) {
#pragma unroll
    for (int r = 0; r < 8; ++r) {
      const int m = m0 + r + rowoff;
      const int n = n0 + j * 16 + col0;
      const float v = acc[j][r];
      if (mode == 0) {
        const int b = m / S, s = m % S;
        const int h = n >> 6, hd = n & 63;
        outH[((size_t)(b * 16 + h) * S + s) * 64 + hd] = (half_t)v;
      } else if (mode == 1) {
        const int b = m / S, s = m % S;
        const int h = n >> 6, hd = n & 63;
        outH[((size_t)(b * 16 + h) * 64 + hd) * S + s] = (half_t)v;
      } else {
        outF[(size_t)m * N + n] = v + bias[n];
      }
    }
  }
}

// ---------------------------------------------------------------------------
// Flash-style attention, one wave per (b, h, 16-row q tile).
// Reference semantics: dots = (q·k)/8 * pad * tril  (masked logits are 0,
// NOT -inf, and participate in softmax) -> full-length online softmax.
__global__ __launch_bounds__(32)
void mha_attn(const half_t* __restrict__ Qh, const half_t* __restrict__ Kh,
              const half_t* __restrict__ Vt, const float* __restrict__ pmask,
              half_t* __restrict__ Oh, int S) {
  __shared__ half_t pt[16 * 64];                   // P tile, row-major [16][64]
  const int lane = threadIdx.x & 31;
  const int q0 = blockIdx.x * 16;
  const int bh = blockIdx.y;                       // b*16 + h
  const int b  = bh >> 4;
  const int h  = bh & 15;

  const half_t* Q  = Qh + (size_t)bh * S * 64;
  const half_t* Kp = Kh + (size_t)bh * S * 64;
  const half_t* Vp = Vt + (size_t)bh * 64 * S;
  const float*  pm = pmask + (size_t)b * S;

  const v16h qa0 = load_a_frag(Q + (size_t)q0 * 64 + 0,  64, lane);
  const v16h qa1 = load_a_frag(Q + (size_t)q0 * 64 + 32, 64, lane);

  v8f o[4];
#pragma unroll
  for (int j = 0; j < 4; ++j) o[j] = (v8f){0.f,0.f,0.f,0.f,0.f,0.f,0.f,0.f};
  float rm[8], rl[8];
#pragma unroll
  for (int r = 0; r < 8; ++r) { rm[r] = -__builtin_inff(); rl[r] = 0.f; }

  const int col    = lane & 15;
  const int rowoff = (lane >> 4) << 3;
  const float sc = 0.125f;                         // 1/sqrt(64)

  for (int k0 = 0; k0 < S; k0 += 64) {
    // ---- scores: S_tile[16 x 64] = Q(16x64) · K_tileᵀ -------------------
    v8f s4[4];
#pragma unroll
    for (int j = 0; j < 4; ++j) {
      v8f c = (v8f){0.f,0.f,0.f,0.f,0.f,0.f,0.f,0.f};
      v16h b0 = load_b_frag(Kp + (size_t)(k0 + j * 16) * 64 + 0,  64, lane);
      v16h b1 = load_b_frag(Kp + (size_t)(k0 + j * 16) * 64 + 32, 64, lane);
      c = WMMA_F16(qa0, b0, c);
      c = WMMA_F16(qa1, b1, c);
      s4[j] = c;
    }
    if (k0 + 64 < S)
      __builtin_prefetch(Kp + (size_t)(k0 + 64 + col) * 64, 0, 0);
    // ---- scale + multiplicative padding/causal masks --------------------
#pragma unroll
    for (int j = 0; j < 4; ++j) {
      const int kc = k0 + j * 16 + col;
      const float pv = pm[kc];
#pragma unroll
      for (int r = 0; r < 8; ++r) {
        const int qr = q0 + r + rowoff;
        float v = s4[j][r] * sc * pv;
        s4[j][r] = (kc <= qr) ? v : 0.0f;
      }
    }
    // ---- online softmax update ------------------------------------------
#pragma unroll
    for (int r = 0; r < 8; ++r) {
      float v = fmaxf(fmaxf(s4[0][r], s4[1][r]), fmaxf(s4[2][r], s4[3][r]));
      v = fmaxf(v, __shfl_xor(v, 1, 32));
      v = fmaxf(v, __shfl_xor(v, 2, 32));
      v = fmaxf(v, __shfl_xor(v, 4, 32));
      v = fmaxf(v, __shfl_xor(v, 8, 32));
      const float nm = fmaxf(rm[r], v);
      const float alpha = __expf(rm[r] - nm);
      rm[r] = nm;
      float ps = 0.f;
#pragma unroll
      for (int j = 0; j < 4; ++j) {
        const float p = __expf(s4[j][r] - nm);
        s4[j][r] = p;
        ps += p;
      }
      ps += __shfl_xor(ps, 1, 32);
      ps += __shfl_xor(ps, 2, 32);
      ps += __shfl_xor(ps, 4, 32);
      ps += __shfl_xor(ps, 8, 32);
      rl[r] = rl[r] * alpha + ps;
#pragma unroll
      for (int j = 0; j < 4; ++j) o[j][r] *= alpha;
    }
    // ---- C-layout -> A-layout via LDS -----------------------------------
#pragma unroll
    for (int j = 0; j < 4; ++j)
#pragma unroll
      for (int r = 0; r < 8; ++r)
        pt[(r + rowoff) * 64 + j * 16 + col] = (half_t)s4[j][r];
    __syncthreads();
    // ---- O += P(16x64) · V(64x64) ---------------------------------------
#pragma unroll
    for (int c = 0; c < 2; ++c) {
      v16h pa = load_a_frag(pt + c * 32, 64, lane);
#pragma unroll
      for (int j = 0; j < 4; ++j) {
        v16h vb = load_b_frag(Vp + (size_t)(j * 16) * S + k0 + c * 32, S, lane);
        o[j] = WMMA_F16(pa, vb, o[j]);
      }
    }
    __syncthreads();
  }

  // ---- normalize + store into O [B,S,D] f16 -----------------------------
#pragma unroll
  for (int j = 0; j < 4; ++j)
#pragma unroll
    for (int r = 0; r < 8; ++r) {
      const int qr = q0 + r + rowoff;
      const float val = o[j][r] / rl[r];
      Oh[((size_t)b * S + qr) * 1024 + h * 64 + j * 16 + col] = (half_t)val;
    }
}

// ---------------------------------------------------------------------------
extern "C" void kernel_launch(void* const* d_in, const int* in_sizes, int n_in,
                              void* d_out, int out_size, void* d_ws, size_t ws_size,
                              hipStream_t stream) {
  const float* x1      = (const float*)d_in[0];
  const float* x2      = (const float*)d_in[1];
  const float* x2_mask = (const float*)d_in[2];
  const float* Wq      = (const float*)d_in[3];
  const float* Wk      = (const float*)d_in[4];
  const float* Wv      = (const float*)d_in[5];
  const float* Wo      = (const float*)d_in[6];
  const float* bo      = (const float*)d_in[7];
  float* out = (float*)d_out;

  const int B = 2, S = 2048, D = 1024;
  const int M = B * S;                              // 4096
  const size_t MSZ = (size_t)M * D;                 // 4 Mi elements
  const size_t WSZ = (size_t)D * D;                 // 1 Mi elements

  half_t* p   = (half_t*)d_ws;
  half_t* x1h = p; p += MSZ;
  half_t* x2h = p; p += MSZ;
  half_t* Wqh = p; p += WSZ;
  half_t* Wkh = p; p += WSZ;
  half_t* Wvh = p; p += WSZ;
  half_t* Woh = p; p += WSZ;
  half_t* Qh  = p; p += MSZ;                        // [B,H,S,HD]
  half_t* Kh  = p; p += MSZ;                        // [B,H,S,HD]
  half_t* Vth = p; p += MSZ;                        // [B,H,HD,S]
  half_t* Ohb = p; p += MSZ;                        // [B,S,D]

  // fp32 -> f16 staging
  mha_cvt_f32_f16<<<1024, 256, 0, stream>>>(x1, x1h, (int)MSZ);
  mha_cvt_f32_f16<<<1024, 256, 0, stream>>>(x2, x2h, (int)MSZ);
  mha_cvt_f32_f16<<<512,  256, 0, stream>>>(Wq, Wqh, (int)WSZ);
  mha_cvt_f32_f16<<<512,  256, 0, stream>>>(Wk, Wkh, (int)WSZ);
  mha_cvt_f32_f16<<<512,  256, 0, stream>>>(Wv, Wvh, (int)WSZ);
  mha_cvt_f32_f16<<<512,  256, 0, stream>>>(Wo, Woh, (int)WSZ);

  dim3 ggrid(M / 128, D / 64);
  // Q = x1·Wqᵀ, K = x2·Wkᵀ  -> [B,H,S,HD];  V = x2·Wvᵀ -> [B,H,HD,S]
  mha_gemm_nt<<<ggrid, 256, 0, stream>>>(x1h, Wqh, M, D, D, S, 0, Qh,  nullptr, nullptr);
  mha_gemm_nt<<<ggrid, 256, 0, stream>>>(x2h, Wkh, M, D, D, S, 0, Kh,  nullptr, nullptr);
  mha_gemm_nt<<<ggrid, 256, 0, stream>>>(x2h, Wvh, M, D, D, S, 1, Vth, nullptr, nullptr);

  // attention
  dim3 agrid(S / 16, B * 16);
  mha_attn<<<agrid, 32, 0, stream>>>(Qh, Kh, Vth, x2_mask, Ohb, S);

  // output projection: out = O·Woᵀ + bo  (fp32)
  mha_gemm_nt<<<ggrid, 256, 0, stream>>>(Ohb, Woh, M, D, D, S, 2, nullptr, out, bo);
}
